// GraphAttention_10720238370878
// MI455X (gfx1250) — matrix operand order repaired and checked by reading, not compiled, not run on hardware
//
#include <hip/hip_runtime.h>
#include <hip/hip_bf16.h>

#define Nn      6144
#define IN_DIM  128
#define HEADS   4
#define ODIM    64
#define CD      256          // HEADS*ODIM
#define NEG     0.2f

typedef _Float16 v16h __attribute__((ext_vector_type(16)));
typedef _Float16 h8v  __attribute__((ext_vector_type(8)));
typedef float    v8f  __attribute__((ext_vector_type(8)));

// ---------------------------------------------------------------------------
// K1: h = input @ W  (fp32), write hT[c][n] in f16 (transposed, for WMMA B),
//     and ciT[h][n] = h[n,h,:]·a_i[h],  cjT[h][n] = h[n,h,:]·a_j[h]
// grid: N/16 blocks, 256 threads (thread t = output column c)
// ---------------------------------------------------------------------------
__global__ __launch_bounds__(256) void gat_proj_kernel(
    const float* __restrict__ input, const float* __restrict__ W,
    const float* __restrict__ a_i, const float* __restrict__ a_j,
    _Float16* __restrict__ hT, float* __restrict__ ciT, float* __restrict__ cjT)
{
    __shared__ float in_lds[16 * IN_DIM];   // 8 KB
    __shared__ float h_lds[16 * CD];        // 16 KB

    const int tid = threadIdx.x;
    const int n0  = blockIdx.x * 16;

    #pragma unroll
    for (int i = 0; i < 8; ++i) {
        int idx = tid + i * 256;                 // 0..2047
        int r = idx >> 7, k = idx & 127;
        in_lds[idx] = input[(size_t)(n0 + r) * IN_DIM + k];
    }
    __syncthreads();

    float acc[16];
    #pragma unroll
    for (int r = 0; r < 16; ++r) acc[r] = 0.0f;
    for (int k = 0; k < IN_DIM; ++k) {
        float wv = W[(size_t)k * CD + tid];
        #pragma unroll
        for (int r = 0; r < 16; ++r)
            acc[r] = __builtin_fmaf(in_lds[r * IN_DIM + k], wv, acc[r]);
    }

    // transposed f16 column chunk: 16 contiguous values at hT[c*N + n0]
    union PK { _Float16 h[2]; unsigned int u; };
    unsigned int up[8];
    #pragma unroll
    for (int j = 0; j < 8; ++j) {
        PK p;
        p.h[0] = (_Float16)acc[2 * j];
        p.h[1] = (_Float16)acc[2 * j + 1];
        up[j] = p.u;
    }
    uint4* dst = (uint4*)(hT + (size_t)tid * Nn + n0);
    dst[0] = make_uint4(up[0], up[1], up[2], up[3]);
    dst[1] = make_uint4(up[4], up[5], up[6], up[7]);

    #pragma unroll
    for (int r = 0; r < 16; ++r) h_lds[r * CD + tid] = acc[r];
    __syncthreads();

    if (tid < 64) {
        int r  = tid >> 2;
        int hh = tid & 3;
        float ci = 0.0f, cj = 0.0f;
        #pragma unroll 8
        for (int d = 0; d < ODIM; ++d) {
            float hv = h_lds[r * CD + hh * ODIM + d];
            ci = __builtin_fmaf(hv, a_i[hh * ODIM + d], ci);
            cj = __builtin_fmaf(hv, a_j[hh * ODIM + d], cj);
        }
        ciT[(size_t)hh * Nn + n0 + r] = ci;
        cjT[(size_t)hh * Nn + n0 + r] = cj;
    }
}

// ---------------------------------------------------------------------------
// K2: per-row max only.  M[n,h] = max(0, max_m s_masked) — any M >= true max
// keeps exp(s-M) <= 1 (f16-safe); the softmax ratio is M-invariant, and the
// denominator is now accumulated inside K3 with the same M.
// grid: N blocks, 256 threads, int4-vectorized adj loads.
// ---------------------------------------------------------------------------
__global__ __launch_bounds__(256) void gat_max_kernel(
    const int* __restrict__ adj, const float* __restrict__ ciT,
    const float* __restrict__ cjT, float* __restrict__ Mrow)
{
    __shared__ float red[4 * 256];           // 4 KB
    const int tid = threadIdx.x;
    const int n   = blockIdx.x;
    const int* rowp = adj + (size_t)n * Nn;

    float cin[4];
    #pragma unroll
    for (int h = 0; h < 4; ++h) cin[h] = ciT[(size_t)h * Nn + n];

    float mx[4] = {0.f, 0.f, 0.f, 0.f};      // 0 is always a safe lower bound
    for (int m = tid * 4; m < Nn; m += 1024) {
        const int4 av = *(const int4*)(rowp + m);
        const int avm[4] = {av.x, av.y, av.z, av.w};
        #pragma unroll
        for (int h = 0; h < 4; ++h) {
            const float4 cj4 = *(const float4*)&cjT[(size_t)h * Nn + m];
            const float cjv[4] = {cj4.x, cj4.y, cj4.z, cj4.w};
            #pragma unroll
            for (int e = 0; e < 4; ++e) {
                float s = cin[h] + cjv[e];
                s = (s >= 0.0f) ? s : NEG * s;
                s = avm[e] ? s : 0.0f;
                mx[h] = fmaxf(mx[h], s);
            }
        }
    }
    #pragma unroll
    for (int h = 0; h < 4; ++h) red[h * 256 + tid] = mx[h];
    __syncthreads();
    for (int off = 128; off > 0; off >>= 1) {
        if (tid < off) {
            #pragma unroll
            for (int h = 0; h < 4; ++h)
                red[h * 256 + tid] = fmaxf(red[h * 256 + tid], red[h * 256 + tid + off]);
        }
        __syncthreads();
    }
    if (tid == 0) {
        #pragma unroll
        for (int h = 0; h < 4; ++h)
            Mrow[(size_t)h * Nn + n] = red[h * 256];
    }
}

// ---------------------------------------------------------------------------
// K3: out[n, h*64+c] = (1/L[n]) * sum_m p[n,m] * hT[h*64+c][m] + bias,
// with L accumulated in-pass:  L = sum_m (adj ? exp(s-M) : exp(-M)).
// Block = (16-row tile, head), 128 threads = 4 waves (wave w -> col-tile w).
// 64-wide m-steps: block produces 16x64 p-tile once into LDS (f16, plain
// [row][k] order), each wave issues 2x v_wmma_f32_16x16x32_f16 per step.
// grid: (N/16, HEADS)
// ---------------------------------------------------------------------------
__global__ __launch_bounds__(128) void gat_attn_kernel(
    const int* __restrict__ adj, const _Float16* __restrict__ hT,
    const float* __restrict__ ciT, const float* __restrict__ cjT,
    const float* __restrict__ Mrow, const float* __restrict__ bias,
    float* __restrict__ out)
{
    __shared__ float    cjh[Nn];             // 24 KB: cj for this head
    __shared__ _Float16 pA[2][16 * 64];      // 4 KB: double-buffered p tile
    __shared__ float    sred[128];           // denom partials

    const int tid  = threadIdx.x;
    const int r0   = blockIdx.x * 16;
    const int head = blockIdx.y;

    for (int i = tid; i < Nn; i += 128)
        cjh[i] = cjT[(size_t)head * Nn + i];

    // ---- producer role: row rr (tid>>3), k-octet kq (tid&7) ----
    const int rr = tid >> 3;
    const int kq = tid & 7;
    const float ci_r  = ciT[(size_t)head * Nn + r0 + rr];
    const float M_r   = Mrow[(size_t)head * Nn + r0 + rr];
    const float enegM = __expf(-M_r);        // denom term for adj==0 entries
    const int* adjrow = adj + (size_t)(r0 + rr) * Nn + kq * 8;
    float sum_p = 0.0f;

    // ---- consumer role (per wave) ----
    const int lane  = tid & 31;
    const int c0    = (tid >> 5) * 16;                 // col tile within head
    const int crow  = lane & 15;                       // A row, B/C column for this lane
    const int bcol  = head * ODIM + c0 + crow;         // global column of hT
    const int koff  = (lane < 16) ? 0 : 16;            // unused (B uses kbase like A)
    const int kbase = (lane < 16) ? 0 : 8;             // A: K base per half-wave
    const _Float16* hcol = hT + (size_t)bcol * Nn;

    v8f acc = {};
    __syncthreads();

    for (int m0 = 0; m0 < Nn; m0 += 64) {
        const int buf = (m0 >> 6) & 1;

        // produce p[rr][kq*8 .. kq*8+7] (branch-free: exp feeds denom always)
        const int4 a0 = *(const int4*)(adjrow + m0);
        const int4 a1 = *(const int4*)(adjrow + m0 + 4);
        const float4 cva = *(const float4*)&cjh[m0 + kq * 8];
        const float4 cvb = *(const float4*)&cjh[m0 + kq * 8 + 4];
        const int   avm[8] = {a0.x, a0.y, a0.z, a0.w, a1.x, a1.y, a1.z, a1.w};
        const float cjv[8] = {cva.x, cva.y, cva.z, cva.w,
                              cvb.x, cvb.y, cvb.z, cvb.w};
        union { _Float16 h[8]; uint4 u; } pk;
        #pragma unroll
        for (int j = 0; j < 8; ++j) {
            float s = ci_r + cjv[j];
            s = (s >= 0.0f) ? s : NEG * s;
            const float p = __expf(s - M_r);           // always live
            sum_p += avm[j] ? p : enegM;               // denom (masked -> exp(-M))
            pk.h[j] = (_Float16)(avm[j] ? p : 0.0f);   // numerator weight
        }
        *(uint4*)&pA[buf][rr * 64 + kq * 8] = pk.u;
        __syncthreads();

        // consume: two 16x16x32 WMMAs over K-chunks [m0, m0+32), [m0+32, m0+64)
        #pragma unroll
        for (int c = 0; c < 2; ++c) {
            h8v lo = *(const h8v*)&pA[buf][crow * 64 + c * 32 + kbase];
            h8v hi = *(const h8v*)&pA[buf][crow * 64 + c * 32 + 16 + kbase];
            v16h a = __builtin_shufflevector(lo, hi, 0, 1, 2, 3, 4, 5, 6, 7,
                                                     8, 9, 10, 11, 12, 13, 14, 15);
            h8v blo = *(const h8v*)(hcol + m0 + c * 32 + kbase);
            h8v bhi = *(const h8v*)(hcol + m0 + c * 32 + 16 + kbase);
            v16h b = __builtin_shufflevector(blo, bhi, 0, 1, 2, 3, 4, 5, 6, 7,
                                                       8, 9, 10, 11, 12, 13, 14, 15);
            acc = __builtin_amdgcn_wmma_f32_16x16x32_f16(
                      false, a, false, b, (short)0, acc, false, false);
        }
        // single barrier per step is safe with double-buffered pA
    }

    // reduce denominators: 8 kq-partials per row -> 1/L
    __syncthreads();
    sred[rr * 8 + kq] = sum_p;
    __syncthreads();
    if (tid < 16) {
        float t = 0.0f;
        #pragma unroll
        for (int j = 0; j < 8; ++j) t += sred[tid * 8 + j];
        sred[tid * 8] = 1.0f / t;
    }
    __syncthreads();

    // epilogue: C layout — VGPR i: lanes 0-15 -> M=i, lanes 16-31 -> M=i+8
    const int colg = head * ODIM + c0 + crow;
    const float bv = bias[colg];
    const int mofs = (lane < 16) ? 0 : 8;
    #pragma unroll
    for (int i = 0; i < 8; ++i) {
        const int rloc = i + mofs;
        const float scale = sred[rloc * 8];
        out[(size_t)(r0 + rloc) * CD + colg] = acc[i] * scale + bv;
    }
}

// ---------------------------------------------------------------------------
extern "C" void kernel_launch(void* const* d_in, const int* in_sizes, int n_in,
                              void* d_out, int out_size, void* d_ws, size_t ws_size,
                              hipStream_t stream) {
    const float* input = (const float*)d_in[0];
    const int*   adj   = (const int*)d_in[1];
    const float* W     = (const float*)d_in[2];
    const float* a_i   = (const float*)d_in[3];
    const float* a_j   = (const float*)d_in[4];
    const float* bias  = (const float*)d_in[5];
    float* out = (float*)d_out;

    char* ws = (char*)d_ws;
    _Float16* hT  = (_Float16*)ws;                         // 256*6144*2 = 3 MB
    float*    ciT = (float*)(ws + (size_t)CD * Nn * 2);
    float*    cjT = ciT + (size_t)HEADS * Nn;
    float*    Mrw = cjT + (size_t)HEADS * Nn;

    gat_proj_kernel<<<Nn / 16, 256, 0, stream>>>(input, W, a_i, a_j, hT, ciT, cjT);
    gat_max_kernel<<<Nn, 256, 0, stream>>>(adj, ciT, cjT, Mrw);
    gat_attn_kernel<<<dim3(Nn / 16, HEADS), 128, 0, stream>>>(
        adj, hT, ciT, cjT, Mrw, bias, out);
}
